// MMN_55748675502962
// MI455X (gfx1250) — compile-verified, matrix operand order checked by reading, not run on hardware
//
#include <hip/hip_runtime.h>
#include <cstdint>

// ---------------------------------------------------------------------------
// Problem constants (from reference)
// ---------------------------------------------------------------------------
constexpr int LL  = 3;
constexpr int BB  = 2;
constexpr int CC  = 2048;
constexpr int HW  = 3600;          // 60*60
constexpr int CH  = 256;
constexpr int KK  = LL * CC;       // 6144 folded reduction dim
constexpr int QP  = 3712;          // padded q rows (58*64, also 29*128)
constexpr int SP  = 3712;          // padded s rows
constexpr float TEMP_  = 20.0f;
constexpr float ATT_WT = 0.3f;
constexpr float EPS_   = 1e-12f;

typedef __attribute__((ext_vector_type(16))) _Float16 v16h;
typedef __attribute__((ext_vector_type(8)))  float    v8f;
typedef unsigned int v4u __attribute__((ext_vector_type(4)));
typedef int          v4i __attribute__((ext_vector_type(4)));
typedef int          v8i __attribute__((ext_vector_type(8)));

// CDNA5 data-movement feature switches
#define USE_ASYNC_AB 1
#if __has_builtin(__builtin_amdgcn_tensor_load_to_lds)
#define USE_TDM_V 1
#else
#define USE_TDM_V 0
#endif

// ---------------------------------------------------------------------------
// Fragment loader (CDNA5 ISA 7.12.2 16-bit layouts, wave32).
// Works for A (row=M) and for B when the tile is stored [N][K] row-major.
// Pairs are contiguous -> lowers to 2x ds_load_b128 per fragment.
// ---------------------------------------------------------------------------
__device__ __forceinline__ v16h frag_rowK(const _Float16* base, int row0, int ld,
                                          int k0, int lane) {
  const int m  = row0 + (lane & 15);
  const int kb = k0 + ((lane & 16) ? 8 : 0);
  const _Float16* p = base + m * ld + kb;
  v16h f;
#pragma unroll
  for (int j = 0; j < 8; ++j) {
    const int k = 2 * j + ((j >= 4) ? 8 : 0);
    f[2 * j]     = p[k];
    f[2 * j + 1] = p[k + 1];
  }
  return f;
}

// ---------------------------------------------------------------------------
// Kernel 1: channel-wise inverse L2 norms (corr_weight folded into query side)
// ---------------------------------------------------------------------------
__global__ __launch_bounds__(256) void norm_kernel(
    const float* __restrict__ fq, const float* __restrict__ fs,
    const float* __restrict__ w, float* __restrict__ invA,
    float* __restrict__ invB) {
  const int lb  = blockIdx.y;                // 0..L*B-1
  const int l   = lb / BB;
  const int pos = blockIdx.x * 256 + threadIdx.x;
  if (pos >= HW) return;
  const size_t base = ((size_t)lb * CC) * HW + pos;
  float sq = 0.f, ss = 0.f;
  for (int c = 0; c < CC; ++c) {
    const float a = fq[base + (size_t)c * HW];
    const float b = fs[base + (size_t)c * HW];
    sq += a * a;
    ss += b * b;
  }
  invA[(size_t)lb * HW + pos] = w[l] / fmaxf(sqrtf(sq), EPS_);
  invB[(size_t)lb * HW + pos] = 1.0f / fmaxf(sqrtf(ss), EPS_);
}

// ---------------------------------------------------------------------------
// Kernel 2: normalize + transpose-pack to f16, K-contiguous (LDS transpose)
// ---------------------------------------------------------------------------
__global__ __launch_bounds__(256) void pack_kernel(
    const float* __restrict__ fq, const float* __restrict__ fs,
    const float* __restrict__ invA, const float* __restrict__ invB,
    _Float16* __restrict__ Apack, _Float16* __restrict__ Bpack) {
  __shared__ float tile[64][65];
  const int p0 = blockIdx.x * 64;            // 58 tiles -> 3712
  const int c0 = blockIdx.y * 64;            // 32 tiles -> 2048
  const int lb = blockIdx.z;                 // l*B+b
  const int l  = lb / BB;
  const int b  = lb % BB;
  const int tid = threadIdx.x;

#pragma unroll
  for (int phase = 0; phase < 2; ++phase) {
    const float*    src = phase ? fs : fq;
    const float*    inv = phase ? invB : invA;
    _Float16*       dst = phase ? Bpack : Apack;
#pragma unroll
    for (int i = 0; i < 16; ++i) {
      const int idx = tid + i * 256;
      const int cc = idx >> 6, pp = idx & 63;
      const int pos = p0 + pp;
      float v = 0.f;
      if (pos < HW)
        v = src[(((size_t)lb * CC) + (c0 + cc)) * HW + pos];
      tile[cc][pp] = v;
    }
    __syncthreads();
#pragma unroll
    for (int i = 0; i < 16; ++i) {
      const int idx = tid + i * 256;
      const int pp = idx >> 6, cc = idx & 63;
      const int pos = p0 + pp;
      const float sc = (pos < HW) ? inv[(size_t)lb * HW + pos] : 0.f;
      dst[((size_t)b * QP + pos) * KK + l * CC + (c0 + cc)] =
          (_Float16)(tile[cc][pp] * sc);
    }
    __syncthreads();
  }
}

// ---------------------------------------------------------------------------
// Kernel 3: pack V = f_s as [b][c][s_pad] f16 (native layout; s padded to SP).
// ---------------------------------------------------------------------------
__global__ __launch_bounds__(256) void packV_kernel(const float* __restrict__ fsv,
                                                    _Float16* __restrict__ Vpack) {
  const size_t n = (size_t)BB * CH * SP;
  const size_t i = (size_t)blockIdx.x * 256 + threadIdx.x;
  if (i >= n) return;
  const int    s   = (int)(i % SP);
  const size_t row = i / SP;                 // b*CH + c
  Vpack[i] = (s < HW) ? (_Float16)fsv[row * HW + s] : (_Float16)0.f;
}

// ---------------------------------------------------------------------------
// Kernel 4: fused red = A.B^T (K=6144), online softmax over s, O = P.V^T.
// Block = 128 threads = 4 wave32; wave owns 16 q rows (BM=64), BN=128 s-chunk.
// A/B tiles: double-buffered global_load_async_to_lds_b128 (ASYNCcnt).
// V tile:    one 2-D TDM tensor_load_to_lds per s-chunk (TENSORcnt).
// B/V fragments: register double-buffered so ds_load latency overlaps WMMA.
// ---------------------------------------------------------------------------
__global__ __launch_bounds__(128) void attn_kernel(
    const _Float16* __restrict__ Apack, const _Float16* __restrict__ Bpack,
    const _Float16* __restrict__ Vpack, const float* __restrict__ f_q,
    float* __restrict__ out_fq, float* __restrict__ out_att) {
  __shared__ __align__(16) _Float16 lA[2 * 64 * 64];   // 16 KB double-buffered
  __shared__ __align__(16) _Float16 lB[2 * 128 * 64];  // 32 KB double-buffered
  __shared__ __align__(16) _Float16 lP[64 * 128];      // 16 KB [qrow][scol]
  __shared__ __align__(16) _Float16 lV[256 * 128];     // 64 KB [c][srow]
  __shared__ __align__(16) float    stO[64 * 257];     // 64.25 KB epilogue stage

  const int tid  = threadIdx.x;
  const int lane = tid & 31;
  const int wq   = tid >> 5;                 // wave id 0..3
  const int lo   = lane & 15;
  const int hi   = (lane >> 4) & 1;
  const int qBase = blockIdx.x * 64;
  const int b     = blockIdx.y;

  const _Float16* Ab = Apack + ((size_t)b * QP + qBase) * KK;
  const _Float16* Bb = Bpack + (size_t)b * SP * KK;
  const _Float16* Vb = Vpack + (size_t)b * CH * SP;

  const v8f zero8 = {0.f, 0.f, 0.f, 0.f, 0.f, 0.f, 0.f, 0.f};
  float m_r[8], l_r[8];
  v8f accO[16];
#pragma unroll
  for (int v = 0; v < 8; ++v) { m_r[v] = -1e30f; l_r[v] = 0.f; }
#pragma unroll
  for (int ct = 0; ct < 16; ++ct) accO[ct] = zero8;

  constexpr int KTILES = KK / 64;            // 96
  constexpr int S_ITERS = (HW + 127) / 128;  // 29

  for (int sc = 0; sc < S_ITERS; ++sc) {
    const int sBase = sc * 128;
    const _Float16* Bs = Bb + (size_t)sBase * KK;

    // ---- V tile: 2-D TDM (256 rows of 128 halves, row stride SP) ----
#if USE_TDM_V
    if (wq == 0) {
      const uint64_t ga = (uint64_t)(uintptr_t)(Vb + sBase);
      const uint32_t la = (uint32_t)(uintptr_t)lV;
      // D# group0: count=1 | lds_addr | global_addr[56:0] | type=2
      v4u g0 = {1u, la, (unsigned)ga,
                (unsigned)((ga >> 32) & 0x1FFFFFFu) | (2u << 30)};
      // group1: data_size=2B; tensor_dim0=3712; tensor_dim1=256;
      //         tile_dim0=128; tile_dim1=256; dim0_stride=3712
      v8i g1 = {(int)0x00010000u,            // wg_mask=0, data_size=1(2B)
                (int)((unsigned)SP << 16),   // tensor_dim0 lo16 << 16
                (int)((unsigned)CH << 16),   // dim0 hi=0 | tensor_dim1 lo16
                (int)(128u << 16),           // dim1 hi=0 | tile_dim0=128
                (int)CH,                     // tile_dim1=256 | tile_dim2=0
                (int)SP,                     // tensor_dim0_stride lo32
                0, 0};
      v4i z4 = {0, 0, 0, 0};
#if defined(__clang_major__) && (__clang_major__ >= 23)
      v8i z8 = {0, 0, 0, 0, 0, 0, 0, 0};
      __builtin_amdgcn_tensor_load_to_lds(g0, g1, z4, z4, z8, 0);
#else
      __builtin_amdgcn_tensor_load_to_lds(g0, g1, z4, z4, 0);
#endif
    }
#else
    {
      float4* dst = (float4*)lV;
#pragma unroll
      for (int i = 0; i < 32; ++i) {
        const int idx = tid + i * 128;       // 4096 float4
        const int r = idx >> 4, s8 = idx & 15;     // row c, 8-half group
        dst[r * 16 + s8] = ((const float4*)(Vb + (size_t)r * SP + sBase))[s8];
      }
    }
#endif

    // stage-issue helper: 12 async b128 per wave per stage
    const auto issueAB = [&](int kt) {
      const _Float16* gA = Ab + kt * 64;
      const _Float16* gB = Bs + kt * 64;
      _Float16* dA = lA + (kt & 1) * (64 * 64);
      _Float16* dB = lB + (kt & 1) * (128 * 64);
#if USE_ASYNC_AB
#pragma unroll
      for (int i = 0; i < 4; ++i) {          // A: 64 rows x 64 halves
        const int idx = tid + i * 128;
        const int r = idx >> 3, k8 = idx & 7;
        const uint32_t la = (uint32_t)(uintptr_t)(dA + r * 64 + k8 * 8);
        const uint64_t ga = (uint64_t)(uintptr_t)(gA + (size_t)r * KK + k8 * 8);
        asm volatile("global_load_async_to_lds_b128 %0, %1, off"
                     :: "v"(la), "v"(ga) : "memory");
      }
#pragma unroll
      for (int i = 0; i < 8; ++i) {          // B: 128 rows x 64 halves
        const int idx = tid + i * 128;
        const int r = idx >> 3, k8 = idx & 7;
        const uint32_t la = (uint32_t)(uintptr_t)(dB + r * 64 + k8 * 8);
        const uint64_t ga = (uint64_t)(uintptr_t)(gB + (size_t)r * KK + k8 * 8);
        asm volatile("global_load_async_to_lds_b128 %0, %1, off"
                     :: "v"(la), "v"(ga) : "memory");
      }
#else
#pragma unroll
      for (int i = 0; i < 4; ++i) {
        const int idx = tid + i * 128;
        const int r = idx >> 3, k8 = idx & 7;
        ((float4*)dA)[r * 8 + k8] =
            ((const float4*)gA)[(size_t)r * (KK / 8) + k8];
      }
#pragma unroll
      for (int i = 0; i < 8; ++i) {
        const int idx = tid + i * 128;
        const int r = idx >> 3, k8 = idx & 7;
        ((float4*)dB)[r * 8 + k8] =
            ((const float4*)gB)[(size_t)r * (KK / 8) + k8];
      }
#endif
    };

    // ---- S = A . B^T over K, KC=64 per stage, double-buffered ----
    v8f accS[8];
#pragma unroll
    for (int t = 0; t < 8; ++t) accS[t] = zero8;

    issueAB(0);
    for (int kt = 0; kt < KTILES; ++kt) {
      __syncthreads();                        // all done computing stage kt-1
      if (kt + 1 < KTILES) {
        __builtin_prefetch((const void*)(Ab + (kt + 2) * 64), 0, 0);
        __builtin_prefetch((const void*)(Bs + (kt + 2) * 64), 0, 0);
        issueAB(kt + 1);
#if USE_ASYNC_AB
        asm volatile("s_wait_asynccnt 12" ::: "memory");  // drain stage kt only
#endif
      } else {
#if USE_ASYNC_AB
        asm volatile("s_wait_asynccnt 0" ::: "memory");
#endif
      }
      __syncthreads();                        // stage kt visible to all waves

      const _Float16* cA = lA + (kt & 1) * (64 * 64);
      const _Float16* cB = lB + (kt & 1) * (128 * 64);
      const v16h fa0 = frag_rowK(cA, wq * 16, 64, 0, lane);
      const v16h fa1 = frag_rowK(cA, wq * 16, 64, 32, lane);
      // register double-buffer: load fragment t+1 while WMMA t executes
      v16h fb0 = frag_rowK(cB, 0, 64, 0, lane);
      v16h fb1 = frag_rowK(cB, 0, 64, 32, lane);
#pragma unroll
      for (int t = 0; t < 8; ++t) {
        v16h nb0, nb1;
        if (t < 7) {
          nb0 = frag_rowK(cB, (t + 1) * 16, 64, 0, lane);
          nb1 = frag_rowK(cB, (t + 1) * 16, 64, 32, lane);
        }
        accS[t] = __builtin_amdgcn_wmma_f32_16x16x32_f16(
            false, fa0, false, fb0, (short)0, accS[t], false, false);
        accS[t] = __builtin_amdgcn_wmma_f32_16x16x32_f16(
            false, fa1, false, fb1, (short)0, accS[t], false, false);
        if (t < 7) { fb0 = nb0; fb1 = nb1; }
      }
    }

    // ---- online softmax over this 128-wide s chunk ----
    bool colOK[8];
#pragma unroll
    for (int t = 0; t < 8; ++t) colOK[t] = (sBase + t * 16 + lo) < HW;

    __syncthreads();                          // lP free (prev PV done)
#pragma unroll
    for (int v = 0; v < 8; ++v) {
      float mx = m_r[v];
#pragma unroll
      for (int t = 0; t < 8; ++t)
        if (colOK[t]) mx = fmaxf(mx, accS[t][v] * TEMP_);
#pragma unroll
      for (int off = 1; off < 16; off <<= 1)
        mx = fmaxf(mx, __shfl_xor(mx, off, 16));
      const float ef = __expf(m_r[v] - mx);
      float rs = 0.f;
      const int row = wq * 16 + hi * 8 + v;
#pragma unroll
      for (int t = 0; t < 8; ++t) {
        const float p = colOK[t] ? __expf(accS[t][v] * TEMP_ - mx) : 0.f;
        rs += p;
        lP[row * 128 + t * 16 + lo] = (_Float16)p;
      }
#pragma unroll
      for (int off = 1; off < 16; off <<= 1) rs += __shfl_xor(rs, off, 16);
      l_r[v] = l_r[v] * ef + rs;
      m_r[v] = mx;
#pragma unroll
      for (int ct = 0; ct < 16; ++ct) accO[ct][v] *= ef;
    }
#if USE_TDM_V
    if (wq == 0) {                            // drain TDM before lV is read
#if __has_builtin(__builtin_amdgcn_s_wait_tensorcnt)
      __builtin_amdgcn_s_wait_tensorcnt(0);
#else
      asm volatile("s_wait_tensorcnt 0" ::: "memory");
#endif
    }
#endif
    __syncthreads();

    // ---- O += P . V^T  (K = 128 chunk cols; lV is [c][s] -> row-contig) ----
#pragma unroll
    for (int ks = 0; ks < 4; ++ks) {
      const v16h fp = frag_rowK(lP, wq * 16, 128, ks * 32, lane);
      v16h fv = frag_rowK(lV, 0, 128, ks * 32, lane);
#pragma unroll
      for (int ct = 0; ct < 16; ++ct) {
        v16h nv;
        if (ct < 15) nv = frag_rowK(lV, (ct + 1) * 16, 128, ks * 32, lane);
        accO[ct] = __builtin_amdgcn_wmma_f32_16x16x32_f16(
            false, fp, false, fv, (short)0, accO[ct], false, false);
        if (ct < 15) fv = nv;
      }
    }
    __syncthreads();                          // before next chunk's lV reload
  }

  // ---- epilogue: stage O via LDS for coalesced q-fast stores ----
#pragma unroll
  for (int v = 0; v < 8; ++v) {
    const float linv = 1.0f / l_r[v];
    const int row = wq * 16 + hi * 8 + v;
#pragma unroll
    for (int ct = 0; ct < 16; ++ct)
      stO[row * 257 + ct * 16 + lo] = accO[ct][v] * linv;
  }
  __syncthreads();
  for (int i = 0; i < 128; ++i) {
    const int idx = tid + i * 128;            // 64 q x 256 c
    const int c  = idx >> 6;
    const int qq = idx & 63;
    const int q  = qBase + qq;
    if (q < HW) {
      const float att = stO[qq * 257 + c];
      const size_t oi = ((size_t)b * CH + c) * HW + q;
      out_att[oi] = att;
      out_fq[oi]  = (f_q[oi] + ATT_WT * att) * (1.0f / (1.0f + ATT_WT));
    }
  }
}

// ---------------------------------------------------------------------------
// Launch
// ---------------------------------------------------------------------------
extern "C" void kernel_launch(void* const* d_in, const int* in_sizes, int n_in,
                              void* d_out, int out_size, void* d_ws, size_t ws_size,
                              hipStream_t stream) {
  const float* fq_feats = (const float*)d_in[0];
  const float* fs_feats = (const float*)d_in[1];
  const float* f_q      = (const float*)d_in[2];
  const float* f_s      = (const float*)d_in[3];
  const float* w        = (const float*)d_in[4];

  float* out_fq  = (float*)d_out;
  float* out_att = out_fq + (size_t)BB * CH * HW;

  // workspace layout (needs ~187 MB)
  const size_t A_BYTES = (size_t)BB * QP * KK * sizeof(_Float16);  // 91,226,112
  const size_t V_BYTES = (size_t)BB * CH * SP * sizeof(_Float16);  //  3,801,088
  const size_t I_BYTES = (((size_t)LL * BB * HW * 4) + 255) & ~(size_t)255;
  char* ws = (char*)d_ws;
  _Float16* Apack = (_Float16*)ws;
  _Float16* Bpack = (_Float16*)(ws + A_BYTES);
  _Float16* Vpack = (_Float16*)(ws + 2 * A_BYTES);
  float*    invA  = (float*)(ws + 2 * A_BYTES + V_BYTES);
  float*    invB  = (float*)(ws + 2 * A_BYTES + V_BYTES + I_BYTES);

  norm_kernel<<<dim3((HW + 255) / 256, LL * BB), 256, 0, stream>>>(
      fq_feats, fs_feats, w, invA, invB);
  pack_kernel<<<dim3(QP / 64, CC / 64, LL * BB), 256, 0, stream>>>(
      fq_feats, fs_feats, invA, invB, Apack, Bpack);
  const size_t nV = (size_t)BB * CH * SP;
  packV_kernel<<<dim3((unsigned)((nV + 255) / 256)), 256, 0, stream>>>(f_s, Vpack);
  attn_kernel<<<dim3((HW + 63) / 64, BB), 128, 0, stream>>>(
      Apack, Bpack, Vpack, f_q, out_fq, out_att);
}
